// GATv2_41180146434474
// MI455X (gfx1250) — matrix operand order repaired and checked by reading, not compiled, not run on hardware
//
#include <hip/hip_runtime.h>
#include <hip/hip_bf16.h>
#include <math.h>

typedef __attribute__((ext_vector_type(16))) __bf16 v16bf;
typedef __attribute__((ext_vector_type(8)))  float  v8f;

#define CH        256      // H*C per conv output
#define HG_STRIDE 320      // pooled(256) + mol(64)

static __device__ __forceinline__ float siluf(float x) { return x / (1.f + expf(-x)); }

static __device__ __forceinline__ __bf16 f2bf(float f) {
    unsigned u = __float_as_uint(f);
    unsigned r = (u + 0x7fffu + ((u >> 16) & 1u)) >> 16;   // round-to-nearest-even
    unsigned short h = (unsigned short)r;
    return __builtin_bit_cast(__bf16, h);
}

static __device__ __forceinline__ void atomicMaxF(float* addr, float val) {
    int* ia = (int*)addr;
    int old = __float_as_int(*addr);
    while (val > __int_as_float(old)) {
        int assumed = old;
        old = atomicCAS(ia, assumed, __float_as_int(val));
        if (old == assumed) break;
    }
}

// ---------------------------------------------------------------- utilities
__global__ void k_fill_f32(float* p, float v, size_t n) {
    size_t i = (size_t)blockIdx.x * blockDim.x + threadIdx.x;
    if (i < n) p[i] = v;
}

__global__ void k_f32_to_bf16(const float* __restrict__ in, __bf16* __restrict__ out, size_t n) {
    size_t i = (size_t)blockIdx.x * blockDim.x + threadIdx.x;
    if (i < n) out[i] = f2bf(in[i]);
}

__global__ void k_addbias(float* __restrict__ x, const float* __restrict__ b, size_t n, int C) {
    size_t i = (size_t)blockIdx.x * blockDim.x + threadIdx.x;
    if (i < n) x[i] += b[i % C];
}

// -------------------------------------------------------- WMMA bf16 GEMM
// C[M,Nout] = A[M,K] @ W[Nout,K]^T + bias   (A,W bf16 row-major, C f32)
// one wave -> 16x64 tile; 8 waves/block. All four B fragments are loaded into
// distinct registers before the WMMA chain so the compiler can issue one load
// clause and retire WMMAs under partial loadcnt waits (load/MMA overlap).
__global__ __launch_bounds__(256) void k_gemm_bf16_wmma(
    const __bf16* __restrict__ A, const __bf16* __restrict__ W,
    const float* __restrict__ bias, float* __restrict__ C,
    int Mtiles, int K, int Nout)
{
    const int ngroups = Nout >> 6;
    const int wave = blockIdx.x * (blockDim.x >> 5) + (threadIdx.x >> 5);
    const int lane = threadIdx.x & 31;
    const int mt = wave / ngroups;
    const int ng = wave - mt * ngroups;
    if (mt >= Mtiles) return;
    const int l16 = lane & 15;
    const int hi  = lane >> 4;
    const size_t arow = (size_t)(mt * 16 + l16) * K;

    // per-lane base pointers for the four B columns (advance by 32 elems per kb)
    const __bf16* bptr[4];
    #pragma unroll
    for (int t = 0; t < 4; ++t) {
        const int col = (ng * 4 + t) * 16 + l16;
        bptr[t] = W + (size_t)col * K + hi * 16;
    }
    const __bf16* aptr = A + arow + hi * 16;

    v8f acc[4] = {};
    union Frag { v16bf v; uint4 q[2]; };

    for (int kb = 0; kb < K; kb += 32) {
        Frag a;
        {
            const uint4* ap = (const uint4*)(aptr + kb);
            a.q[0] = ap[0];      // K = kb + hi*16 + [0..7]
            a.q[1] = ap[1];      // K = kb + hi*16 + 8 + [0..7]
        }
        Frag b0, b1, b2, b3;
        {
            const uint4* p0 = (const uint4*)(bptr[0] + kb);
            const uint4* p1 = (const uint4*)(bptr[1] + kb);
            const uint4* p2 = (const uint4*)(bptr[2] + kb);
            const uint4* p3 = (const uint4*)(bptr[3] + kb);
            b0.q[0] = p0[0]; b0.q[1] = p0[1];
            b1.q[0] = p1[0]; b1.q[1] = p1[1];
            b2.q[0] = p2[0]; b2.q[1] = p2[1];
            b3.q[0] = p3[0]; b3.q[1] = p3[1];
        }
        acc[0] = __builtin_amdgcn_wmma_f32_16x16x32_bf16(false, a.v, false, b0.v, (short)0, acc[0], false, false);
        acc[1] = __builtin_amdgcn_wmma_f32_16x16x32_bf16(false, a.v, false, b1.v, (short)0, acc[1], false, false);
        acc[2] = __builtin_amdgcn_wmma_f32_16x16x32_bf16(false, a.v, false, b2.v, (short)0, acc[2], false, false);
        acc[3] = __builtin_amdgcn_wmma_f32_16x16x32_bf16(false, a.v, false, b3.v, (short)0, acc[3], false, false);
    }
    #pragma unroll
    for (int t = 0; t < 4; ++t) {
        const int ncol = (ng * 4 + t) * 16 + l16;
        const float bv = bias[ncol];
        const int mbase = mt * 16 + hi * 8;
        #pragma unroll
        for (int v = 0; v < 8; ++v)
            C[(size_t)(mbase + v) * Nout + ncol] = acc[t][v] + bv;
    }
}

// ------------------------------------------------------------ edge kernels
// pass 1: logit[e,h] = sum_c att[h,c]*lrelu(xl[s,h,c]+xr[d,h,c]); atomic max per dst
__global__ void k_edge_logit_max(const int* __restrict__ ei,
    const float* __restrict__ xl, const float* __restrict__ xr,
    const float* __restrict__ att, float* __restrict__ logit,
    float* __restrict__ m, int E, int N)
{
    int idx = blockIdx.x * blockDim.x + threadIdx.x;
    int total = (E + N) * 4;
    if (idx >= total) return;
    int e = idx >> 2, h = idx & 3;
    int s, d;
    if (e < E) { s = ei[e]; d = ei[E + e]; } else { s = d = e - E; }
    const float4* a4 = (const float4*)(xl + (size_t)s * CH + h * 64);
    const float4* b4 = (const float4*)(xr + (size_t)d * CH + h * 64);
    const float4* t4 = (const float4*)(att + h * 64);
    float acc = 0.f;
    #pragma unroll
    for (int j = 0; j < 16; ++j) {
        float4 a = a4[j], b = b4[j], t = t4[j];
        float vx = a.x + b.x, vy = a.y + b.y, vz = a.z + b.z, vw = a.w + b.w;
        vx = vx > 0.f ? vx : 0.2f * vx;
        vy = vy > 0.f ? vy : 0.2f * vy;
        vz = vz > 0.f ? vz : 0.2f * vz;
        vw = vw > 0.f ? vw : 0.2f * vw;
        acc += vx * t.x + vy * t.y + vz * t.z + vw * t.w;
    }
    logit[idx] = acc;
    atomicMaxF(&m[d * 4 + h], acc);
}

// pass 2: num = exp(logit - m[dst]); atomic add into den
__global__ void k_edge_softmax_num(const int* __restrict__ ei,
    float* __restrict__ logit, const float* __restrict__ m,
    float* __restrict__ den, int E, int N)
{
    int idx = blockIdx.x * blockDim.x + threadIdx.x;
    int total = (E + N) * 4;
    if (idx >= total) return;
    int e = idx >> 2, h = idx & 3;
    int d = (e < E) ? ei[E + e] : e - E;
    float num = expf(logit[idx] - m[d * 4 + h]);
    logit[idx] = num;
    atomicAdd(&den[d * 4 + h], num);
}

// pass 3: out[d] += xl[s] * alpha
__global__ void k_edge_aggregate(const int* __restrict__ ei,
    const float* __restrict__ xl, const float* __restrict__ num,
    const float* __restrict__ den, float* __restrict__ hout, int E, int N)
{
    size_t idx = (size_t)blockIdx.x * blockDim.x + threadIdx.x;
    size_t total = (size_t)(E + N) * 64;
    if (idx >= total) return;
    int e = (int)(idx >> 6), q = (int)(idx & 63);
    int s, d;
    if (e < E) { s = ei[e]; d = ei[E + e]; } else { s = d = e - E; }
    int h = q >> 4;  // q*4 channels -> head = (q*4)/64
    float alpha = num[(size_t)e * 4 + h] / (den[d * 4 + h] + 1e-16f);
    float4 v = ((const float4*)xl)[(size_t)s * 64 + q];
    float* o = hout + (size_t)d * CH + q * 4;
    atomicAdd(o + 0, v.x * alpha);
    atomicAdd(o + 1, v.y * alpha);
    atomicAdd(o + 2, v.z * alpha);
    atomicAdd(o + 3, v.w * alpha);
}

// --------------------------------------------------------- stats / BN / act
__global__ void k_colstats(const float* __restrict__ x, float* __restrict__ stats,
                           int Nrows, int C)
{
    int c = threadIdx.x;                 // blockDim.x == C
    float s = 0.f, sq = 0.f;
    for (int r = blockIdx.x; r < Nrows; r += gridDim.x) {
        float v = x[(size_t)r * C + c];
        s += v; sq += v * v;
    }
    atomicAdd(&stats[c], s);
    atomicAdd(&stats[C + c], sq);
}

__global__ void k_bn_silu_bf16(const float* __restrict__ x, const float* __restrict__ stats,
    const float* __restrict__ gamma, const float* __restrict__ beta,
    __bf16* __restrict__ out, size_t n, int C, float invN)
{
    size_t i = (size_t)blockIdx.x * blockDim.x + threadIdx.x;
    if (i >= n) return;
    int c = (int)(i % C);
    float mu = stats[c] * invN;
    float var = stats[C + c] * invN - mu * mu;
    float y = gamma[c] * (x[i] - mu) * rsqrtf(var + 1e-5f) + beta[c];
    out[i] = f2bf(siluf(y));
}

__global__ void k_bn_silu_f32(const float* __restrict__ x, const float* __restrict__ stats,
    const float* __restrict__ gamma, const float* __restrict__ beta,
    float* __restrict__ out, size_t n, int C, float invN)
{
    size_t i = (size_t)blockIdx.x * blockDim.x + threadIdx.x;
    if (i >= n) return;
    int c = (int)(i % C);
    float mu = stats[c] * invN;
    float var = stats[C + c] * invN - mu * mu;
    float y = gamma[c] * (x[i] - mu) * rsqrtf(var + 1e-5f) + beta[c];
    out[i] = siluf(y);
}

__global__ void k_silu_bf16(const float* __restrict__ x, const float* __restrict__ bias,
                            __bf16* __restrict__ out, size_t n, int C)
{
    size_t i = (size_t)blockIdx.x * blockDim.x + threadIdx.x;
    if (i >= n) return;
    out[i] = f2bf(siluf(x[i] + bias[i % C]));
}

__global__ void k_silu_pool(const float* __restrict__ x, const float* __restrict__ bias,
                            const int* __restrict__ batch, float* __restrict__ hg, size_t n)
{
    size_t i = (size_t)blockIdx.x * blockDim.x + threadIdx.x;
    if (i >= n) return;
    int r = (int)(i >> 8), c = (int)(i & 255);
    float v = siluf(x[i] + bias[c]);
    atomicAdd(&hg[(size_t)batch[r] * HG_STRIDE + c], v);
}

// ------------------------------------------------------------- small linear
__global__ void k_linear(const float* __restrict__ in, const float* __restrict__ W,
    const float* __restrict__ b, float* __restrict__ out,
    int M, int K, int Nout, int inStride, int outStride, int act)
{
    int idx = blockIdx.x * blockDim.x + threadIdx.x;
    if (idx >= M * Nout) return;
    int r = idx / Nout, o = idx - r * Nout;
    const float* ip = in + (size_t)r * inStride;
    const float* wp = W + (size_t)o * K;
    float acc = b[o];
    for (int k = 0; k < K; ++k) acc += ip[k] * wp[k];
    if (act) acc = siluf(acc);
    out[(size_t)r * outStride + o] = acc;
}

// =========================================================== host launcher
extern "C" void kernel_launch(void* const* d_in, const int* in_sizes, int n_in,
                              void* d_out, int out_size, void* d_ws, size_t ws_size,
                              hipStream_t stream)
{
    if (n_in < 38) return;
    const float* x     = (const float*)d_in[0];
    const int*   ei    = (const int*)d_in[1];
    const int*   batch = (const int*)d_in[2];
    const float* mol   = (const float*)d_in[3];
    const float *Wl[3], *bl[3], *Wr[3], *br[3], *attp[3], *bias[3];
    for (int l = 0; l < 3; ++l) {
        int base = 4 + l * 6;
        Wl[l]   = (const float*)d_in[base + 0];
        bl[l]   = (const float*)d_in[base + 1];
        Wr[l]   = (const float*)d_in[base + 2];
        br[l]   = (const float*)d_in[base + 3];
        attp[l] = (const float*)d_in[base + 4];
        bias[l] = (const float*)d_in[base + 5];
    }
    const float* g_gc = (const float*)d_in[22];
    const float* b_gc = (const float*)d_in[23];
    const float* W0m  = (const float*)d_in[24];
    const float* b0m  = (const float*)d_in[25];
    const float* W1m  = (const float*)d_in[26];
    const float* b1m  = (const float*)d_in[27];
    const float* g_m  = (const float*)d_in[28];
    const float* b_m  = (const float*)d_in[29];
    const float* Wf[4] = {(const float*)d_in[30], (const float*)d_in[32],
                          (const float*)d_in[34], (const float*)d_in[36]};
    const float* bf[4] = {(const float*)d_in[31], (const float*)d_in[33],
                          (const float*)d_in[35], (const float*)d_in[37]};

    const int N = in_sizes[0] / 64;        // 50000
    const int E = in_sizes[1] / 2;         // 800000
    const int G = in_sizes[3] / 200;       // 256
    const int Mpad = (N + 15) & ~15;       // 50000
    const int Mtiles = Mpad / 16;
    const int Et = E + N;

    // ------- workspace carve-out (256B aligned)
    char* wsb = (char*)d_ws;
    size_t off = 0;
    auto carve = [&](size_t bytes) -> void* {
        off = (off + 255) & ~(size_t)255;
        void* p = wsb + off; off += bytes; return p;
    };
    __bf16* wlb[3], *wrb[3];
    const int Kin[3] = {64, 256, 256};
    for (int l = 0; l < 3; ++l) {
        wlb[l] = (__bf16*)carve((size_t)256 * Kin[l] * 2);
        wrb[l] = (__bf16*)carve((size_t)256 * Kin[l] * 2);
    }
    __bf16* hb   = (__bf16*)carve((size_t)Mpad * CH * 2);
    float* xl    = (float*)carve((size_t)Mpad * CH * 4);
    float* xr    = (float*)carve((size_t)Mpad * CH * 4);
    float* hout  = (float*)carve((size_t)Mpad * CH * 4);
    float* logit = (float*)carve((size_t)Et * 4 * 4);
    float* mbuf  = (float*)carve((size_t)N * 4 * 4);
    float* den   = (float*)carve((size_t)N * 4 * 4);
    float* stats = (float*)carve(512 * 4);
    float* hg    = (float*)carve((size_t)G * HG_STRIDE * 4);
    float* zm    = (float*)carve((size_t)G * 64 * 4);
    float* hm1   = (float*)carve((size_t)G * 64 * 4);
    float* z1    = (float*)carve((size_t)G * 512 * 4);
    float* z2    = (float*)carve((size_t)G * 512 * 4);
    float* z3    = (float*)carve((size_t)G * 256 * 4);
    (void)ws_size;

    auto blk = [](size_t n, int t) { return (unsigned)((n + t - 1) / t); };

    // ------- weight + input conversion to bf16
    for (int l = 0; l < 3; ++l) {
        size_t wn = (size_t)256 * Kin[l];
        k_f32_to_bf16<<<blk(wn, 256), 256, 0, stream>>>(Wl[l], wlb[l], wn);
        k_f32_to_bf16<<<blk(wn, 256), 256, 0, stream>>>(Wr[l], wrb[l], wn);
    }
    k_f32_to_bf16<<<blk((size_t)N * 64, 256), 256, 0, stream>>>(x, hb, (size_t)N * 64);
    k_fill_f32<<<blk((size_t)G * HG_STRIDE, 256), 256, 0, stream>>>(hg, 0.f, (size_t)G * HG_STRIDE);

    // ------- 3 GATv2 convolutions
    for (int l = 0; l < 3; ++l) {
        int K = Kin[l];
        unsigned gblocks = (unsigned)((Mtiles * 4 + 7) / 8);
        k_gemm_bf16_wmma<<<gblocks, 256, 0, stream>>>(hb, wlb[l], bl[l], xl, Mtiles, K, CH);
        k_gemm_bf16_wmma<<<gblocks, 256, 0, stream>>>(hb, wrb[l], br[l], xr, Mtiles, K, CH);

        k_fill_f32<<<blk((size_t)N * 4, 256), 256, 0, stream>>>(mbuf, -3.0e38f, (size_t)N * 4);
        k_fill_f32<<<blk((size_t)N * 4, 256), 256, 0, stream>>>(den, 0.f, (size_t)N * 4);
        k_fill_f32<<<blk((size_t)Mpad * CH, 256), 256, 0, stream>>>(hout, 0.f, (size_t)Mpad * CH);

        k_edge_logit_max<<<blk((size_t)Et * 4, 256), 256, 0, stream>>>(ei, xl, xr, attp[l], logit, mbuf, E, N);
        k_edge_softmax_num<<<blk((size_t)Et * 4, 256), 256, 0, stream>>>(ei, logit, mbuf, den, E, N);
        k_edge_aggregate<<<blk((size_t)Et * 64, 256), 256, 0, stream>>>(ei, xl, logit, den, hout, E, N);

        size_t hn = (size_t)N * CH;
        if (l == 0) {
            k_addbias<<<blk(hn, 256), 256, 0, stream>>>(hout, bias[0], hn, CH);
            k_fill_f32<<<2, 256, 0, stream>>>(stats, 0.f, 512);
            k_colstats<<<512, CH, 0, stream>>>(hout, stats, N, CH);
            k_bn_silu_bf16<<<blk(hn, 256), 256, 0, stream>>>(hout, stats, g_gc, b_gc, hb, hn, CH, 1.f / N);
        } else if (l == 1) {
            k_silu_bf16<<<blk(hn, 256), 256, 0, stream>>>(hout, bias[1], hb, hn, CH);
        } else {
            k_silu_pool<<<blk(hn, 256), 256, 0, stream>>>(hout, bias[2], batch, hg, hn);
        }
    }

    // ------- molecular branch -> hg columns [256..319]
    k_linear<<<blk((size_t)G * 64, 256), 256, 0, stream>>>(mol, W0m, b0m, zm, G, 200, 64, 200, 64, 0);
    k_fill_f32<<<1, 128, 0, stream>>>(stats, 0.f, 128);
    k_colstats<<<64, 64, 0, stream>>>(zm, stats, G, 64);
    k_bn_silu_f32<<<blk((size_t)G * 64, 256), 256, 0, stream>>>(zm, stats, g_m, b_m, hm1, (size_t)G * 64, 64, 1.f / G);
    k_linear<<<blk((size_t)G * 64, 256), 256, 0, stream>>>(hm1, W1m, b1m, hg + 256, G, 64, 64, 64, HG_STRIDE, 1);

    // ------- final MLP head
    k_linear<<<blk((size_t)G * 512, 256), 256, 0, stream>>>(hg, Wf[0], bf[0], z1, G, HG_STRIDE, 512, HG_STRIDE, 512, 1);
    k_linear<<<blk((size_t)G * 512, 256), 256, 0, stream>>>(z1, Wf[1], bf[1], z2, G, 512, 512, 512, 512, 1);
    k_linear<<<blk((size_t)G * 256, 256), 256, 0, stream>>>(z2, Wf[2], bf[2], z3, G, 512, 256, 512, 256, 1);
    k_linear<<<blk((size_t)G, 256), 256, 0, stream>>>(z3, Wf[3], bf[3], (float*)d_out, G, 256, 1, 256, 1, 0);
}
